// Net_27762668601860
// MI455X (gfx1250) — compile-verified, matrix operand order checked by reading, not run on hardware
//
#include <hip/hip_runtime.h>
#include <math.h>

// Problem constants (match reference)
#define B_GR 128
#define NG_  500
#define K1_  250
#define K2_  125
#define H_   128
#define OUT_ 10
#define NNODES (B_GR * NG_)

typedef _Float16 f16;
typedef __attribute__((ext_vector_type(16))) _Float16 v16h;
typedef __attribute__((ext_vector_type(8)))  float    v8f;

__device__ __forceinline__ float selu_f(float x) {
  const float alpha = 1.6732632423543772f;
  const float scale = 1.0507009873554805f;
  return x > 0.f ? scale * x : scale * alpha * (__expf(x) - 1.f);
}

// ---------------------------------------------------------------------------
// Batched WMMA GEMM:  D[b] = ep( op(A[b]) @ B[b] + bias + Cadd )
//   op(A): [M,K]  (TRANSA: stored [K,M]); B: [K,N] row-major.
// CLDA/CLDB are compile-time leading dimensions (-1 = runtime fallback) so
// every strided load uses a constant instruction offset from a base pointer
// that advances by 32*ld once per K-step.  One wave = 4 N-subtiles sharing
// one A fragment (4 back-to-back WMMAs per K-step).
// Out-of-range rows/cols are clamped (their outputs are never stored); the
// K%32 tail uses value masks and may over-read <=31 rows (guard-padded ws).
// ---------------------------------------------------------------------------
template <int TRANSA, int EP, bool HASB, bool HASC, int CLDA, int CLDB>
__global__ __launch_bounds__(128) void wmma_gemm_kernel(
    const float* __restrict__ A, const float* __restrict__ Bm,
    const float* __restrict__ bias, const float* __restrict__ Cadd,
    float* __restrict__ D, int batch, int M, int N, int K,
    long sA, long sB, long sD, int lda_rt, int ldb_rt, int ldd) {
  const int lda = CLDA > 0 ? CLDA : lda_rt;
  const int ldb = CLDB > 0 ? CLDB : ldb_rt;

  const int tilesM  = (M + 15) >> 4;
  const int tilesN  = (N + 15) >> 4;
  const int groupsN = (tilesN + 3) >> 2;                 // 4 N-tiles per wave
  const long tilesPerBatch = (long)tilesM * groupsN;
  const long total = (long)batch * tilesPerBatch;
  const long tile  = (long)blockIdx.x * blockDim.y + threadIdx.y;
  if (tile >= total) return;                             // wave-uniform

  const int b   = (int)(tile / tilesPerBatch);
  const int t   = (int)(tile - (long)b * tilesPerBatch);
  const int tm  = t / groupsN;
  const int tg  = t - tm * groupsN;

  const int lane = threadIdx.x;
  const int lr   = lane & 15;
  const int half = lane >> 4;

  const float* __restrict__ Ab = A  + (long)b * sA;
  const float* __restrict__ Bb = Bm + (long)b * sB;

  const int m_a   = tm * 16 + lr;
  const int m_eff = m_a < M ? m_a : M - 1;               // clamp (safe addr)

  int n_col[4], n_eff[4];
#pragma unroll
  for (int i = 0; i < 4; ++i) {
    n_col[i] = (tg * 4 + i) * 16 + lr;
    n_eff[i] = n_col[i] < N ? n_col[i] : N - 1;
  }

  // per-lane base pointers; inner-loop offsets are compile-time constants
  const float* __restrict__ Ap =
      TRANSA ? (Ab + m_eff + (long)(half * 8) * lda)
             : (Ab + (long)m_eff * lda + half * 8);
  const float* __restrict__ Bp[4];
#pragma unroll
  for (int i = 0; i < 4; ++i) Bp[i] = Bb + n_eff[i] + (long)(half * 8) * ldb;

  v8f acc[4];
  if (HASC) {
    const float* __restrict__ Cb = Cadd + (long)b * sD;
#pragma unroll
    for (int i = 0; i < 4; ++i)
#pragma unroll
      for (int j = 0; j < 8; ++j) {
        int m  = tm * 16 + half * 8 + j;
        int me = m < M ? m : M - 1;
        acc[i][j] = Cb[(long)me * ldd + n_eff[i]];
      }
  } else {
#pragma unroll
    for (int i = 0; i < 4; ++i)
#pragma unroll
      for (int j = 0; j < 8; ++j) acc[i][j] = 0.f;
  }

  const int Kfull = K & ~31;
  int k0 = 0;
  for (; k0 < Kfull; k0 += 32) {                         // branch-free hot loop
    v16h af;
    if (TRANSA == 0 && CLDA > 0 && (CLDA % 4) == 0) {
      const float4* __restrict__ p4 = reinterpret_cast<const float4*>(Ap);
      float4 q0 = p4[0], q1 = p4[1];                     // k = base+0..7
      float4 q2 = p4[4], q3 = p4[5];                     // k = base+16..23
      af[0]  = (f16)q0.x; af[1]  = (f16)q0.y; af[2]  = (f16)q0.z; af[3]  = (f16)q0.w;
      af[4]  = (f16)q1.x; af[5]  = (f16)q1.y; af[6]  = (f16)q1.z; af[7]  = (f16)q1.w;
      af[8]  = (f16)q2.x; af[9]  = (f16)q2.y; af[10] = (f16)q2.z; af[11] = (f16)q2.w;
      af[12] = (f16)q3.x; af[13] = (f16)q3.y; af[14] = (f16)q3.z; af[15] = (f16)q3.w;
    } else if (TRANSA) {
#pragma unroll
      for (int e = 0; e < 8; ++e) {
        af[e]     = (f16)Ap[(long)e * lda];
        af[e + 8] = (f16)Ap[(long)(e + 16) * lda];
      }
    } else {
#pragma unroll
      for (int e = 0; e < 8; ++e) {
        af[e]     = (f16)Ap[e];
        af[e + 8] = (f16)Ap[e + 16];
      }
    }
#pragma unroll
    for (int i = 0; i < 4; ++i) {
      v16h bf;
#pragma unroll
      for (int e = 0; e < 8; ++e) {
        bf[e]     = (f16)Bp[i][(long)e * ldb];
        bf[e + 8] = (f16)Bp[i][(long)(e + 16) * ldb];
      }
      acc[i] = __builtin_amdgcn_wmma_f32_16x16x32_f16(false, af, false, bf,
                                                      (short)0, acc[i], false,
                                                      false);
    }
    Ap += TRANSA ? (long)32 * lda : 32;
#pragma unroll
    for (int i = 0; i < 4; ++i) Bp[i] += (long)32 * ldb;
  }

  if (k0 < K) {                                          // masked K tail
    const int rem = K - k0;                              // 1..31
    v16h af;
#pragma unroll
    for (int e = 0; e < 8; ++e) {
      int ka = half * 8 + e, kb = ka + 16;
      float va = TRANSA ? Ap[(long)e * lda] : Ap[e];
      float vb = TRANSA ? Ap[(long)(e + 16) * lda] : Ap[e + 16];
      af[e]     = (f16)(ka < rem ? va : 0.f);
      af[e + 8] = (f16)(kb < rem ? vb : 0.f);
    }
#pragma unroll
    for (int i = 0; i < 4; ++i) {
      v16h bf;
#pragma unroll
      for (int e = 0; e < 8; ++e) {
        int ka = half * 8 + e, kb = ka + 16;
        float wa = Bp[i][(long)e * ldb];
        float wb = Bp[i][(long)(e + 16) * ldb];
        bf[e]     = (f16)(ka < rem ? wa : 0.f);
        bf[e + 8] = (f16)(kb < rem ? wb : 0.f);
      }
      acc[i] = __builtin_amdgcn_wmma_f32_16x16x32_f16(false, af, false, bf,
                                                      (short)0, acc[i], false,
                                                      false);
    }
  }

  float* __restrict__ Db = D + (long)b * sD;
#pragma unroll
  for (int i = 0; i < 4; ++i) {
    float bv = HASB ? bias[n_eff[i]] : 0.f;
#pragma unroll
    for (int j = 0; j < 8; ++j) {
      int m = tm * 16 + half * 8 + j;
      if (m < M && n_col[i] < N) {
        float v = acc[i][j] + bv;
        if (EP == 1)      v = fmaxf(v, 0.f);
        else if (EP == 2) v = selu_f(v);
        Db[(long)m * ldd + n_col[i]] = v;
      }
    }
  }
}

// ---------------------------------------------------------------------------
// Small helper kernels
// ---------------------------------------------------------------------------
__global__ void fill_zero_kernel(float* p, long n) {
  long i = (long)blockIdx.x * blockDim.x + threadIdx.x;
  if (i < n) p[i] = 0.f;
}

__global__ void scatter_adj_kernel(const int* __restrict__ row,
                                   const int* __restrict__ col,
                                   float* __restrict__ adj, int E) {
  int e = blockIdx.x * blockDim.x + threadIdx.x;
  if (e >= E) return;
  int r = row[e], c = col[e];
  int g  = r / NG_;
  int pr = r - g * NG_;
  int pc = c - g * NG_;
  atomicAdd(&adj[((long)g * NG_ + pr) * NG_ + pc], 1.0f);
}

// dis[b*NG+c] = rsqrt( 1 + sum_r adj[b,r,c] )
__global__ void gcn_deg_dis_kernel(const float* __restrict__ adj,
                                   float* __restrict__ dis) {
  long i = (long)blockIdx.x * blockDim.x + threadIdx.x;
  if (i >= (long)B_GR * NG_) return;
  int b = (int)(i / NG_), c = (int)(i % NG_);
  const float* a = adj + (long)b * NG_ * NG_ + c;
  float s = 0.f;
  for (int r = 0; r < NG_; ++r) s += a[(long)r * NG_];
  dis[i] = rsqrtf(s + 1.0f);
}

__global__ void scale_rows_kernel(const float* __restrict__ h,
                                  const float* __restrict__ dis,
                                  float* __restrict__ hs, long n) {
  long i = (long)blockIdx.x * blockDim.x + threadIdx.x;
  if (i >= n) return;
  hs[i] = h[i] * dis[i / H_];
}

__global__ void gcn_finish_kernel(const float* __restrict__ agg,
                                  const float* __restrict__ hs,
                                  const float* __restrict__ dis,
                                  const float* __restrict__ bias,
                                  float* __restrict__ xd, long n) {
  long i = (long)blockIdx.x * blockDim.x + threadIdx.x;
  if (i >= n) return;
  float v = dis[i / H_] * (agg[i] + hs[i]) + bias[i % H_];
  xd[i] = fmaxf(v, 0.f);
}

// in-place row softmax; one wave (32 lanes) per row
__global__ void softmax_rows_kernel(float* __restrict__ x, long rows, int C) {
  long row = (long)blockIdx.x * blockDim.y + threadIdx.y;
  if (row >= rows) return;
  float* p = x + row * (long)C;
  int lane = threadIdx.x;
  float mx = -3.4e38f;
  for (int j = lane; j < C; j += 32) mx = fmaxf(mx, p[j]);
  for (int o = 16; o > 0; o >>= 1) mx = fmaxf(mx, __shfl_xor(mx, o, 32));
  float sum = 0.f;
  for (int j = lane; j < C; j += 32) sum += __expf(p[j] - mx);
  for (int o = 16; o > 0; o >>= 1) sum += __shfl_xor(sum, o, 32);
  float inv = 1.0f / sum;
  for (int j = lane; j < C; j += 32) p[j] = __expf(p[j] - mx) * inv;
}

__device__ __forceinline__ float block_sum256(float v, float* red) {
  int tid = threadIdx.x;
  red[tid] = v;
  __syncthreads();
  for (int s = 128; s > 0; s >>= 1) {
    if (tid < s) red[tid] += red[tid + s];
    __syncthreads();
  }
  float r = red[0];
  __syncthreads();
  return r;
}

// Per-graph DMoN loss statistics. One 256-thread block per graph.
__global__ void dmon_stats_kernel(const float* __restrict__ s,
                                  const float* __restrict__ adj,
                                  const float* __restrict__ out_adj,
                                  const float* __restrict__ ss,
                                  int n, int k,
                                  float* __restrict__ spec,
                                  float* __restrict__ orth,
                                  float* __restrict__ csn) {
  __shared__ float deg_sh[512];
  __shared__ float ca_sh[256];
  __shared__ float csz_sh[256];
  __shared__ float red[256];
  const int b = blockIdx.x, tid = threadIdx.x, T = 256;
  const float* A = adj + (long)b * n * n;
  const float* S = s + (long)b * n * k;

  for (int m = tid; m < n; m += T) {            // deg = column sums of adj
    float acc = 0.f;
    for (int i = 0; i < n; ++i) acc += A[(long)i * n + m];
    deg_sh[m] = acc;
  }
  __syncthreads();

  float loc = 0.f;
  for (int m = tid; m < n; m += T) loc += deg_sh[m];
  float m2 = 0.5f * block_sum256(loc, red);

  for (int kk = tid; kk < k; kk += T) {         // ca, csz
    float a = 0.f, c = 0.f;
    for (int i = 0; i < n; ++i) {
      float sv = S[(long)i * k + kk];
      a += sv * deg_sh[i];
      c += sv;
    }
    ca_sh[kk] = a;
    csz_sh[kk] = c;
  }
  __syncthreads();

  float l1 = 0.f, l2 = 0.f;
  for (int kk = tid; kk < k; kk += T) {
    l1 += ca_sh[kk] * ca_sh[kk];
    l2 += csz_sh[kk] * csz_sh[kk];
  }
  float sum_ca2 = block_sum256(l1, red);
  float sum_csz2 = block_sum256(l2, red);

  float tdiag = 0.f;
  for (int j = tid; j < k; j += T) tdiag += out_adj[(long)b * k * k + (long)j * k + j];
  float tr_adj = block_sum256(tdiag, red);

  const float* SSb = ss + (long)b * k * k;
  float l3 = 0.f;
  for (long idx = tid; idx < (long)k * k; idx += T) {
    float v = SSb[idx];
    l3 += v * v;
  }
  float ssn = sqrtf(block_sum256(l3, red));

  float inv_ssn = 1.0f / ssn;
  float ik = rsqrtf((float)k);
  float l4 = 0.f;
  for (long idx = tid; idx < (long)k * k; idx += T) {
    int i = (int)(idx / k), j = (int)(idx % k);
    float d = SSb[idx] * inv_ssn - (i == j ? ik : 0.f);
    l4 += d * d;
  }
  float orth2 = block_sum256(l4, red);

  if (tid == 0) {
    float tr_norm = sum_ca2 / (2.0f * m2);
    float trace = tr_adj - tr_norm;
    spec[b] = -trace / (2.0f * m2);
    orth[b] = sqrtf(orth2);
    csn[b] = sqrtf(sum_csz2);
  }
}

__global__ void mean_axis1_kernel(const float* __restrict__ xd,
                                  float* __restrict__ g, int n) {
  int i = blockIdx.x * blockDim.x + threadIdx.x;
  if (i >= B_GR * H_) return;
  int b = i / H_, ch = i % H_;
  const float* p = xd + (long)b * n * H_ + ch;
  float s = 0.f;
  for (int j = 0; j < n; ++j) s += p[(long)j * H_];
  g[i] = s / (float)n;
}

__global__ void logsoftmax_kernel(const float* __restrict__ logits,
                                  float* __restrict__ out) {
  int b = blockIdx.x * blockDim.x + threadIdx.x;
  if (b >= B_GR) return;
  const float* r = logits + (long)b * OUT_;
  float mx = -3.4e38f;
  for (int j = 0; j < OUT_; ++j) mx = fmaxf(mx, r[j]);
  float se = 0.f;
  for (int j = 0; j < OUT_; ++j) se += __expf(r[j] - mx);
  float lse = logf(se);
  for (int j = 0; j < OUT_; ++j) out[(long)b * OUT_ + j] = r[j] - mx - lse;
}

__global__ void aux_final_kernel(const float* __restrict__ stat,
                                 float* __restrict__ out) {
  __shared__ float red[256];
  int tid = threadIdx.x;
  float v[6];
  for (int a = 0; a < 6; ++a)
    v[a] = (tid < B_GR) ? stat[a * B_GR + tid] : 0.f;
  float s0 = block_sum256(v[0], red);  // spec1
  float s1 = block_sum256(v[1], red);  // orth1
  float s2 = block_sum256(v[2], red);  // csn1
  float s3 = block_sum256(v[3], red);  // spec2
  float s4 = block_sum256(v[4], red);  // orth2
  float s5 = block_sum256(v[5], red);  // csn2
  if (tid == 0) {
    const float invB = 1.0f / (float)B_GR;
    float cl1 = sqrtf((float)K1_) * (s2 * invB) / (float)NG_ - 1.0f;
    float cl2 = sqrtf((float)K2_) * (s5 * invB) / (float)K1_ - 1.0f;
    out[B_GR * OUT_] = s0 * invB + s3 * invB + s1 * invB + s4 * invB + cl1 + cl2;
  }
}

// ---------------------------------------------------------------------------
// Host side
// ---------------------------------------------------------------------------
template <int TA, int EP, bool HB, bool HC, int LDA, int LDB>
static void launch_one(hipStream_t st, const float* A, const float* Bm,
                       const float* bias, const float* Cadd, float* D,
                       int batch, int M, int N, int K,
                       long sA, long sB, long sD, int lda, int ldb, int ldd) {
  long tilesM = (M + 15) / 16;
  long groupsN = ((N + 15) / 16 + 3) / 4;
  long tiles = (long)batch * tilesM * groupsN;
  long blocks = (tiles + 3) / 4;
  dim3 grid((unsigned)blocks), block(32, 4);
  wmma_gemm_kernel<TA, EP, HB, HC, LDA, LDB><<<grid, block, 0, st>>>(
      A, Bm, bias, Cadd, D, batch, M, N, K, sA, sB, sD, lda, ldb, ldd);
}

static void launch_gemm(hipStream_t st, const float* A, const float* Bm,
                        const float* bias, const float* Cadd, float* D,
                        int batch, int M, int N, int K,
                        long sA, long sB, long sD,
                        int lda, int ldb, int ldd, int transA, int ep) {
  const bool hb = bias != nullptr, hc = Cadd != nullptr;
#define GEMM_TRY(TA, EPv, HBv, HCv, LDA, LDB)                                  \
  if (transA == (TA) && ep == (EPv) && hb == (HBv) && hc == (HCv) &&           \
      lda == (LDA) && ldb == (LDB)) {                                          \
    launch_one<TA, EPv, HBv, HCv, LDA, LDB>(st, A, Bm, bias, Cadd, D, batch,   \
                                            M, N, K, sA, sB, sD, lda, ldb,     \
                                            ldd);                              \
    return;                                                                    \
  }
  GEMM_TRY(0, 0, false, false, 128, 128)
  GEMM_TRY(0, 0, false, false, 500, 250)
  GEMM_TRY(0, 0, false, false, 250, 128)
  GEMM_TRY(0, 0, false, false, 250, 125)
  GEMM_TRY(0, 0, false, false, 125, 128)
  GEMM_TRY(0, 0, true,  false, 128, 128)
  GEMM_TRY(0, 0, true,  false, 128, 250)
  GEMM_TRY(0, 0, true,  false, 128, 125)
  GEMM_TRY(0, 0, true,  false, 128, 10)
  GEMM_TRY(1, 0, false, false, 500, 128)
  GEMM_TRY(1, 0, false, false, 250, 250)
  GEMM_TRY(1, 0, false, false, 125, 125)
  GEMM_TRY(1, 2, false, false, 250, 128)
  GEMM_TRY(1, 2, false, false, 125, 128)
  GEMM_TRY(0, 1, false, true,  128, 128)
  GEMM_TRY(0, 0, false, true,  128, 128)
  GEMM_TRY(0, 1, true,  false, 128, 128)
#undef GEMM_TRY
  // generic fallback (not expected to be hit)
  launch_one<0, 0, false, false, -1, -1>(st, A, Bm, bias, Cadd, D, batch, M, N,
                                         K, sA, sB, sD, lda, ldb, ldd);
}

static inline unsigned gblk(long n, int t) { return (unsigned)((n + t - 1) / t); }

extern "C" void kernel_launch(void* const* d_in, const int* in_sizes, int n_in,
                              void* d_out, int out_size, void* d_ws, size_t ws_size,
                              hipStream_t stream) {
  const float* x        = (const float*)d_in[0];
  const int*   eidx     = (const int*)d_in[1];
  const int    E        = in_sizes[1] / 2;
  const int*   row      = eidx;
  const int*   col      = eidx + E;
  const float* gcn_w    = (const float*)d_in[3];
  const float* gcn_b    = (const float*)d_in[4];
  const float* p1_w1    = (const float*)d_in[5];
  const float* p1_b1    = (const float*)d_in[6];
  const float* p1_w2    = (const float*)d_in[7];
  const float* p1_b2    = (const float*)d_in[8];
  const float* c2_wrel  = (const float*)d_in[9];
  const float* c2_brel  = (const float*)d_in[10];
  const float* c2_wroot = (const float*)d_in[11];
  const float* p2_w1    = (const float*)d_in[12];
  const float* p2_b1    = (const float*)d_in[13];
  const float* p2_w2    = (const float*)d_in[14];
  const float* p2_b2    = (const float*)d_in[15];
  const float* c3_wrel  = (const float*)d_in[16];
  const float* c3_brel  = (const float*)d_in[17];
  const float* c3_wroot = (const float*)d_in[18];
  const float* lin1_w   = (const float*)d_in[19];
  const float* lin1_b   = (const float*)d_in[20];
  const float* lin2_w   = (const float*)d_in[21];
  const float* lin2_b   = (const float*)d_in[22];
  float* out = (float*)d_out;

  // ---- workspace layout (floats) ----
  float* ws = (float*)d_ws;
  size_t off = 0;
  auto A_ = [&](size_t n) { float* p = ws + off; off += n; return p; };
  float* ADJ1 = A_((size_t)B_GR * NG_ * NG_);      // 32,000,000 (arena later)
  float* H    = A_((size_t)NNODES * H_);           //  8,192,000
  float* HS   = A_((size_t)NNODES * H_);
  float* AGG  = A_((size_t)NNODES * H_);
  float* XD1  = A_((size_t)NNODES * H_);
  float* T    = A_((size_t)NNODES * H_);
  float* DIS  = A_((size_t)NNODES);
  float* S1   = A_((size_t)B_GR * NG_ * K1_);      // 16,000,000
  float* T1   = A_((size_t)B_GR * NG_ * K1_);
  float* OADJ1= A_((size_t)B_GR * K1_ * K1_);      //  8,000,000
  float* SS1  = A_((size_t)B_GR * K1_ * K1_);
  float* XD2  = A_((size_t)B_GR * K1_ * H_);       //  4,096,000
  float* STAT = A_((size_t)6 * B_GR);
  (void)A_((size_t)32768);                         // guard pad for K-tail reads
  // stage-2 arena aliases ADJ1 (dead after dmon_stats #1); ~1.5M floats slack
  size_t aoff = 0;
  auto AR = [&](size_t n) { float* p = ADJ1 + aoff; aoff += n; return p; };
  float* T2    = AR((size_t)B_GR * K1_ * H_);
  float* XD2B  = AR((size_t)B_GR * K1_ * H_);
  float* TM2   = AR((size_t)B_GR * K1_ * H_);
  float* S2    = AR((size_t)B_GR * K1_ * K2_);
  float* T1B   = AR((size_t)B_GR * K1_ * K2_);
  float* OADJ2 = AR((size_t)B_GR * K2_ * K2_);
  float* SS2   = AR((size_t)B_GR * K2_ * K2_);
  float* XD3   = AR((size_t)B_GR * K2_ * H_);
  float* T3    = AR((size_t)B_GR * K2_ * H_);
  float* XD3B  = AR((size_t)B_GR * K2_ * H_);
  float* G     = AR((size_t)B_GR * H_);
  float* G2    = AR((size_t)B_GR * H_);
  float* LOGI  = AR((size_t)B_GR * OUT_);
  (void)ws_size; (void)n_in; (void)out_size;

  const long adjN = (long)B_GR * NG_ * NG_;
  const long nH   = (long)NNODES * H_;

  // ---- 1. dense adjacency ----
  fill_zero_kernel<<<gblk(adjN, 256), 256, 0, stream>>>(ADJ1, adjN);
  scatter_adj_kernel<<<gblk(E, 256), 256, 0, stream>>>(row, col, ADJ1, E);

  // ---- 2. GCNConv (as dense batched GEMM) ----
  launch_gemm(stream, x, gcn_w, nullptr, nullptr, H, 1, NNODES, H_, H_,
              0, 0, 0, H_, H_, H_, 0, 0);                       // H = x@W
  gcn_deg_dis_kernel<<<gblk((long)NNODES, 256), 256, 0, stream>>>(ADJ1, DIS);
  scale_rows_kernel<<<gblk(nH, 256), 256, 0, stream>>>(H, DIS, HS, nH);
  launch_gemm(stream, ADJ1, HS, nullptr, nullptr, AGG, B_GR, NG_, H_, NG_,
              (long)NG_ * NG_, (long)NG_ * H_, (long)NG_ * H_,
              NG_, H_, H_, 1, 0);                               // AGG = adj^T @ HS
  gcn_finish_kernel<<<gblk(nH, 256), 256, 0, stream>>>(AGG, HS, DIS, gcn_b, XD1, nH);

  // ---- 3. DMoN pooling #1 ----
  launch_gemm(stream, XD1, p1_w1, p1_b1, nullptr, T, B_GR, NG_, H_, H_,
              (long)NG_ * H_, 0, (long)NG_ * H_, H_, H_, H_, 0, 0);
  launch_gemm(stream, T, p1_w2, p1_b2, nullptr, S1, B_GR, NG_, K1_, H_,
              (long)NG_ * H_, 0, (long)NG_ * K1_, H_, K1_, K1_, 0, 0);
  {
    dim3 blk(32, 8);
    softmax_rows_kernel<<<gblk((long)NNODES, 8), blk, 0, stream>>>(S1, NNODES, K1_);
  }
  launch_gemm(stream, S1, XD1, nullptr, nullptr, XD2, B_GR, K1_, H_, NG_,
              (long)NG_ * K1_, (long)NG_ * H_, (long)K1_ * H_,
              K1_, H_, H_, 1, 2);                               // XD2 = selu(s^T x)
  launch_gemm(stream, ADJ1, S1, nullptr, nullptr, T1, B_GR, NG_, K1_, NG_,
              (long)NG_ * NG_, (long)NG_ * K1_, (long)NG_ * K1_,
              NG_, K1_, K1_, 0, 0);                             // T1 = adj@s
  launch_gemm(stream, S1, T1, nullptr, nullptr, OADJ1, B_GR, K1_, K1_, NG_,
              (long)NG_ * K1_, (long)NG_ * K1_, (long)K1_ * K1_,
              K1_, K1_, K1_, 1, 0);                             // out_adj = s^T T1
  launch_gemm(stream, S1, S1, nullptr, nullptr, SS1, B_GR, K1_, K1_, NG_,
              (long)NG_ * K1_, (long)NG_ * K1_, (long)K1_ * K1_,
              K1_, K1_, K1_, 1, 0);                             // ss = s^T s
  dmon_stats_kernel<<<B_GR, 256, 0, stream>>>(S1, ADJ1, OADJ1, SS1, NG_, K1_,
                                              STAT + 0 * B_GR, STAT + 1 * B_GR,
                                              STAT + 2 * B_GR);
  // (ADJ1 dead from here; stage-2 arena aliases it)

  // ---- 4. DenseGraphConv #2 ----
  launch_gemm(stream, OADJ1, XD2, nullptr, nullptr, T2, B_GR, K1_, H_, K1_,
              (long)K1_ * K1_, (long)K1_ * H_, (long)K1_ * H_,
              K1_, H_, H_, 0, 0);                               // T2 = adj@x
  launch_gemm(stream, T2, c2_wrel, c2_brel, nullptr, XD2B, B_GR, K1_, H_, H_,
              (long)K1_ * H_, 0, (long)K1_ * H_, H_, H_, H_, 0, 0);
  launch_gemm(stream, XD2, c2_wroot, nullptr, XD2B, XD2B, B_GR, K1_, H_, H_,
              (long)K1_ * H_, 0, (long)K1_ * H_, H_, H_, H_, 0, 1); // +=, relu

  // ---- 5. DMoN pooling #2 ----
  launch_gemm(stream, XD2B, p2_w1, p2_b1, nullptr, TM2, B_GR, K1_, H_, H_,
              (long)K1_ * H_, 0, (long)K1_ * H_, H_, H_, H_, 0, 0);
  launch_gemm(stream, TM2, p2_w2, p2_b2, nullptr, S2, B_GR, K1_, K2_, H_,
              (long)K1_ * H_, 0, (long)K1_ * K2_, H_, K2_, K2_, 0, 0);
  {
    dim3 blk(32, 8);
    softmax_rows_kernel<<<gblk((long)B_GR * K1_, 8), blk, 0, stream>>>(
        S2, (long)B_GR * K1_, K2_);
  }
  launch_gemm(stream, S2, XD2B, nullptr, nullptr, XD3, B_GR, K2_, H_, K1_,
              (long)K1_ * K2_, (long)K1_ * H_, (long)K2_ * H_,
              K2_, H_, H_, 1, 2);                               // selu(s^T x)
  launch_gemm(stream, OADJ1, S2, nullptr, nullptr, T1B, B_GR, K1_, K2_, K1_,
              (long)K1_ * K1_, (long)K1_ * K2_, (long)K1_ * K2_,
              K1_, K2_, K2_, 0, 0);
  launch_gemm(stream, S2, T1B, nullptr, nullptr, OADJ2, B_GR, K2_, K2_, K1_,
              (long)K1_ * K2_, (long)K1_ * K2_, (long)K2_ * K2_,
              K2_, K2_, K2_, 1, 0);
  launch_gemm(stream, S2, S2, nullptr, nullptr, SS2, B_GR, K2_, K2_, K1_,
              (long)K1_ * K2_, (long)K1_ * K2_, (long)K2_ * K2_,
              K2_, K2_, K2_, 1, 0);
  dmon_stats_kernel<<<B_GR, 256, 0, stream>>>(S2, OADJ1, OADJ2, SS2, K1_, K2_,
                                              STAT + 3 * B_GR, STAT + 4 * B_GR,
                                              STAT + 5 * B_GR);

  // ---- 6. DenseGraphConv #3 (no relu) ----
  launch_gemm(stream, OADJ2, XD3, nullptr, nullptr, T3, B_GR, K2_, H_, K2_,
              (long)K2_ * K2_, (long)K2_ * H_, (long)K2_ * H_,
              K2_, H_, H_, 0, 0);
  launch_gemm(stream, T3, c3_wrel, c3_brel, nullptr, XD3B, B_GR, K2_, H_, H_,
              (long)K2_ * H_, 0, (long)K2_ * H_, H_, H_, H_, 0, 0);
  launch_gemm(stream, XD3, c3_wroot, nullptr, XD3B, XD3B, B_GR, K2_, H_, H_,
              (long)K2_ * H_, 0, (long)K2_ * H_, H_, H_, H_, 0, 0);

  // ---- 7. readout + MLP head ----
  mean_axis1_kernel<<<gblk((long)B_GR * H_, 256), 256, 0, stream>>>(XD3B, G, K2_);
  launch_gemm(stream, G, lin1_w, lin1_b, nullptr, G2, 1, B_GR, H_, H_,
              0, 0, 0, H_, H_, H_, 0, 1);                       // relu
  launch_gemm(stream, G2, lin2_w, lin2_b, nullptr, LOGI, 1, B_GR, OUT_, H_,
              0, 0, 0, H_, OUT_, OUT_, 0, 0);
  logsoftmax_kernel<<<1, 128, 0, stream>>>(LOGI, out);
  aux_final_kernel<<<1, 256, 0, stream>>>(STAT, out);
}
